// DownSample_Trans_47485158425077
// MI455X (gfx1250) — matrix operand order repaired and checked by reading, not compiled, not run on hardware
//
#include <hip/hip_runtime.h>

// ---------------- problem constants ----------------
#define BB   4
#define HH   48
#define WW   48
#define CC   2304
#define EE   2304
#define NN   2304      // H*W
#define NHD  4
#define LNEPS 1e-5f
#define PADW 64        // padded image row stride (elements)
#define PADC 3200      // padded image channel stride = 50 * 64

typedef __bf16 bf16_t;
typedef __attribute__((ext_vector_type(16))) __bf16 bf16x16;
typedef __attribute__((ext_vector_type(8)))  __bf16 bf16x8;
typedef __attribute__((ext_vector_type(8)))  float  f32x8;

// ---------------- WMMA fragment loaders (wave32, 16x16x32 bf16) ----------------
// A 16x32 (MxK): lane holds row m = lane&15; elements e=0..7 -> k0+(lane>>4)*8+e,
// e=8..15 -> k0+16+(lane>>4)*8+(e-8). Two contiguous 16B chunks.
__device__ __forceinline__ bf16x16 load_a_frag(const bf16_t* __restrict__ A, long lda,
                                               int mbase, int k0, int lane) {
  const bf16_t* p = A + (long)(mbase + (lane & 15)) * lda + k0 + ((lane >> 4) << 3);
  bf16x8 lo = *(const bf16x8*)p;
  bf16x8 hi = *(const bf16x8*)(p + 16);
  bf16x16 r;
#pragma unroll
  for (int i = 0; i < 8; ++i) { r[i] = lo[i]; r[i + 8] = hi[i]; }
  return r;
}

// B 32x16 (KxN): lane holds column n = n0+(lane&15); elements e=0..7 -> k0+(lane>>4)*8+e,
// e=8..15 -> k0+16+(lane>>4)*8+(e-8).
__device__ __forceinline__ bf16x16 load_b_frag(const bf16_t* __restrict__ B, long ldb,
                                               int k0, int n0, int lane) {
  const bf16_t* p = B + (long)(k0 + ((lane >> 4) << 3)) * ldb + n0 + (lane & 15);
  bf16x16 r;
#pragma unroll
  for (int i = 0; i < 8; ++i) { r[i] = p[(long)i * ldb]; r[i + 8] = p[(long)(i + 16) * ldb]; }
  return r;
}

#define WMMA_BF16(a, b, c) \
  __builtin_amdgcn_wmma_f32_16x16x32_bf16(false, (a), false, (b), (short)0, (c), false, false)

// ---------------- generic bf16 WMMA GEMM, 32x32 tile per wave ----------------
// D[b][m][n] = alpha * sum_k A[b][m][k]*B[b][k][n]  (+bias_m[m]) (+bias_n[n])
//              (+relpos(m,n,b)) ; optional relu; stores fp32 / bf16 / bf16-transposed.
__global__ __launch_bounds__(256)
void gemm_bf16_wmma(const bf16_t* __restrict__ A, long ldA, long sA,
                    const bf16_t* __restrict__ B, long ldB, long sB,
                    int M, int Nn, int K, float alpha,
                    const float* __restrict__ bias_m,
                    const float* __restrict__ bias_n, int relu,
                    float* __restrict__ outF, bf16_t* __restrict__ outB,
                    long ldD, long sD,
                    bf16_t* __restrict__ outT, long ldT, long sT,
                    const float* __restrict__ rel_table, int use_rel) {
  const int lane = threadIdx.x & 31;
  const int wv   = threadIdx.x >> 5;
  const int n0 = (blockIdx.x * 8 + wv) * 32;
  const int m0 = blockIdx.y * 32;
  const int b  = blockIdx.z;
  if (m0 >= M || n0 >= Nn) return;

  const bf16_t* Ab = A + (long)b * sA;
  const bf16_t* Bb = B + (long)b * sB;

  f32x8 acc[2][2];
#pragma unroll
  for (int i = 0; i < 2; ++i)
#pragma unroll
    for (int j = 0; j < 2; ++j)
#pragma unroll
      for (int r = 0; r < 8; ++r) acc[i][j][r] = 0.f;

  for (int k0 = 0; k0 < K; k0 += 32) {
    bf16x16 a0 = load_a_frag(Ab, ldA, m0,      k0, lane);
    bf16x16 a1 = load_a_frag(Ab, ldA, m0 + 16, k0, lane);
    bf16x16 b0 = load_b_frag(Bb, ldB, k0, n0,      lane);
    bf16x16 b1 = load_b_frag(Bb, ldB, k0, n0 + 16, lane);
    acc[0][0] = WMMA_BF16(a0, b0, acc[0][0]);
    acc[0][1] = WMMA_BF16(a0, b1, acc[0][1]);
    acc[1][0] = WMMA_BF16(a1, b0, acc[1][0]);
    acc[1][1] = WMMA_BF16(a1, b1, acc[1][1]);
  }

  const int lh = lane >> 4, ll = lane & 15;
#pragma unroll
  for (int ms = 0; ms < 2; ++ms)
#pragma unroll
    for (int ns = 0; ns < 2; ++ns)
#pragma unroll
      for (int r = 0; r < 8; ++r) {
        int m = m0 + ms * 16 + r + lh * 8;
        int n = n0 + ns * 16 + ll;
        float v = acc[ms][ns][r] * alpha;
        if (bias_m) v += bias_m[m];
        if (bias_n) v += bias_n[n];
        if (use_rel) {
          int yi = m / WW, xi = m - yi * WW;
          int yj = n / WW, xj = n - yj * WW;
          v += rel_table[((yi - yj + HH - 1) * (2 * HH - 1) + (xi - xj + WW - 1)) * NHD + b];
        }
        if (relu) v = fmaxf(v, 0.f);
        long off = (long)b * sD + (long)m * ldD + n;
        if (outF) outF[off] = v;
        if (outB) outB[off] = (bf16_t)v;
        if (outT) outT[(long)b * sT + (long)n * ldT + m] = (bf16_t)v;
      }
}

// ---------------- conv2 (3x3) as 9 shifted GEMMs + identity residual + relu --------
// B-operand fragments are gathered per element from hpad, separately for each tap.
__global__ __launch_bounds__(256)
void conv2_res_gemm(const bf16_t* __restrict__ w2t,   // [9][Cout][Cin]
                    const bf16_t* __restrict__ hpad,  // [B][C][50*64] zero-padded
                    const float* __restrict__ c2b,
                    const float* __restrict__ idw, const float* __restrict__ idb,
                    const float* __restrict__ xin,    // [B][2304]
                    float* __restrict__ downF, bf16_t* __restrict__ downB) {
  const int lane = threadIdx.x & 31;
  const int wv   = threadIdx.x >> 5;
  const int n0 = (blockIdx.x * 8 + wv) * 32;
  const int m0 = blockIdx.y * 32;
  const int b  = blockIdx.z;
  const bf16_t* hb = hpad + (long)b * CC * PADC;

  int py[2], px[2];
#pragma unroll
  for (int ns = 0; ns < 2; ++ns) { int nb = n0 + ns * 16; py[ns] = nb / WW; px[ns] = nb % WW; }

  f32x8 acc[2][2];
#pragma unroll
  for (int i = 0; i < 2; ++i)
#pragma unroll
    for (int j = 0; j < 2; ++j)
#pragma unroll
      for (int r = 0; r < 8; ++r) acc[i][j][r] = 0.f;

  for (int k0 = 0; k0 < CC; k0 += 32) {
#pragma unroll
    for (int dy = 0; dy < 3; ++dy) {
#pragma unroll
      for (int dx = 0; dx < 3; ++dx) {
        const bf16_t* At = w2t + (long)(dy * 3 + dx) * CC * CC;
        bf16x16 a0 = load_a_frag(At, CC, m0,      k0, lane);
        bf16x16 a1 = load_a_frag(At, CC, m0 + 16, k0, lane);
        bf16x16 b0, b1;
#pragma unroll
        for (int e = 0; e < 16; ++e) {
          const long kk = (long)(k0 + ((lane >> 4) << 3) + (e & 7) + ((e >> 3) << 4)) * PADC;
          b0[e] = hb[kk + (py[0] + dy) * PADW + px[0] + (lane & 15) + dx];
          b1[e] = hb[kk + (py[1] + dy) * PADW + px[1] + (lane & 15) + dx];
        }
        acc[0][0] = WMMA_BF16(a0, b0, acc[0][0]);
        acc[0][1] = WMMA_BF16(a0, b1, acc[0][1]);
        acc[1][0] = WMMA_BF16(a1, b0, acc[1][0]);
        acc[1][1] = WMMA_BF16(a1, b1, acc[1][1]);
      }
    }
  }

  const int lh = lane >> 4, ll = lane & 15;
#pragma unroll
  for (int ms = 0; ms < 2; ++ms)
#pragma unroll
    for (int ns = 0; ns < 2; ++ns)
#pragma unroll
      for (int r = 0; r < 8; ++r) {
        int m = m0 + ms * 16 + r + lh * 8;
        int n = n0 + ns * 16 + ll;
        float v = acc[ms][ns][r] + c2b[m] + idw[m] * xin[(long)b * NN + n] + idb[m];
        v = fmaxf(v, 0.f);
        long off = (long)b * CC * NN + (long)m * NN + n;
        downF[off] = v;
        downB[off] = (bf16_t)v;
      }
}

// ---------------- conv1 (CIN=1) + relu, write zero-padded bf16 image ----------------
__global__ void conv1_relu_pad(const float* __restrict__ x, const float* __restrict__ w,
                               const float* __restrict__ bias, bf16_t* __restrict__ hpad) {
  long idx = (long)blockIdx.x * blockDim.x + threadIdx.x;
  const long total = (long)BB * CC * 2500;
  if (idx >= total) return;
  int r  = (int)(idx % 2500);
  long bc = idx / 2500;
  int yy = r / 50, xx = r % 50;
  long out = bc * PADC + yy * PADW + xx;
  if (yy == 0 || yy == 49 || xx == 0 || xx == 49) { hpad[out] = (bf16_t)0.f; return; }
  int b = (int)(bc / CC), c = (int)(bc % CC);
  int yo = yy - 1, xo = xx - 1;
  float s = bias[c];
#pragma unroll
  for (int ky = 0; ky < 3; ++ky) {
    int iy = yo + ky - 1;
    if ((unsigned)iy >= HH) continue;
#pragma unroll
    for (int kx = 0; kx < 3; ++kx) {
      int ix = xo + kx - 1;
      if ((unsigned)ix >= WW) continue;
      s += x[(long)b * NN + iy * WW + ix] * w[c * 9 + ky * 3 + kx];
    }
  }
  hpad[out] = (bf16_t)fmaxf(s, 0.f);
}

// ---------------- weight repack / convert ----------------
__global__ void repack_conv2(const float* __restrict__ w, bf16_t* __restrict__ o) {
  long i = (long)blockIdx.x * blockDim.x + threadIdx.x;
  if (i >= (long)9 * CC * CC) return;
  int t = (int)(i / ((long)CC * CC));
  long rc = i - (long)t * CC * CC;            // co*C + ci
  o[i] = (bf16_t)w[rc * 9 + t];
}

__global__ void f32_to_bf16(const float* __restrict__ in, bf16_t* __restrict__ out, long n) {
  long i = (long)blockIdx.x * blockDim.x + threadIdx.x;
  if (i < n) out[i] = (bf16_t)in[i];
}

// ---------------- softmax rows (length 2304) ----------------
__global__ __launch_bounds__(256)
void softmax_rows(const float* __restrict__ qk, bf16_t* __restrict__ attn) {
  __shared__ float red[256];
  const float* src = qk + (long)blockIdx.x * EE;
  bf16_t* dst = attn + (long)blockIdx.x * EE;
  int tid = threadIdx.x;
  float mx = -1e30f;
  for (int j = tid; j < EE; j += 256) mx = fmaxf(mx, src[j]);
  red[tid] = mx; __syncthreads();
  for (int s = 128; s > 0; s >>= 1) { if (tid < s) red[tid] = fmaxf(red[tid], red[tid + s]); __syncthreads(); }
  mx = red[0]; __syncthreads();
  float sum = 0.f;
  for (int j = tid; j < EE; j += 256) sum += __expf(src[j] - mx);
  red[tid] = sum; __syncthreads();
  for (int s = 128; s > 0; s >>= 1) { if (tid < s) red[tid] += red[tid + s]; __syncthreads(); }
  float inv = 1.f / red[0];
  for (int j = tid; j < EE; j += 256) dst[j] = (bf16_t)(__expf(src[j] - mx) * inv);
}

// ---------------- LN1: gather-permute ao + down residual, LayerNorm ----------------
__global__ __launch_bounds__(256)
void ln1_residual(const float* __restrict__ ao,   // (B,E,N)
                  const float* __restrict__ down, // (B*C, N)
                  const float* __restrict__ g, const float* __restrict__ be,
                  float* __restrict__ h1f, bf16_t* __restrict__ h1b) {
  __shared__ float buf[NN];
  __shared__ float red[256];
  const int G = blockIdx.x;           // b'*C + c'
  const int e = G >> 2, bb = G & 3;   // ao permute: e*B + b == G
  const float* s1 = ao + ((long)bb * EE + e) * NN;
  const float* s2 = down + (long)G * NN;
  int tid = threadIdx.x;
  float sum = 0.f;
  for (int n = tid; n < NN; n += 256) { float t = s1[n] + s2[n]; buf[n] = t; sum += t; }
  red[tid] = sum; __syncthreads();
  for (int s = 128; s > 0; s >>= 1) { if (tid < s) red[tid] += red[tid + s]; __syncthreads(); }
  float mu = red[0] * (1.f / NN); __syncthreads();
  float vs = 0.f;
  for (int n = tid; n < NN; n += 256) { float d = buf[n] - mu; vs += d * d; }
  red[tid] = vs; __syncthreads();
  for (int s = 128; s > 0; s >>= 1) { if (tid < s) red[tid] += red[tid + s]; __syncthreads(); }
  float inv = rsqrtf(red[0] * (1.f / NN) + LNEPS);
  for (int n = tid; n < NN; n += 256) {
    float y = (buf[n] - mu) * inv * g[n] + be[n];
    h1f[(long)G * NN + n] = y;
    h1b[(long)G * NN + n] = (bf16_t)y;
  }
}

// ---------------- LN2 + residual + fused 2x2 maxpool ----------------
__global__ __launch_bounds__(256)
void ln2_pool(const float* __restrict__ f, const float* __restrict__ h1f,
              const float* __restrict__ g, const float* __restrict__ be,
              float* __restrict__ out, float* __restrict__ pool) {
  __shared__ float buf[NN];
  __shared__ float red[256];
  const int G = blockIdx.x;
  const float* s1 = f + (long)G * NN;
  const float* s2 = h1f + (long)G * NN;
  int tid = threadIdx.x;
  float sum = 0.f;
  for (int n = tid; n < NN; n += 256) { float t = s1[n] + s2[n]; buf[n] = t; sum += t; }
  red[tid] = sum; __syncthreads();
  for (int s = 128; s > 0; s >>= 1) { if (tid < s) red[tid] += red[tid + s]; __syncthreads(); }
  float mu = red[0] * (1.f / NN); __syncthreads();
  float vs = 0.f;
  for (int n = tid; n < NN; n += 256) { float d = buf[n] - mu; vs += d * d; }
  red[tid] = vs; __syncthreads();
  for (int s = 128; s > 0; s >>= 1) { if (tid < s) red[tid] += red[tid + s]; __syncthreads(); }
  float inv = rsqrtf(red[0] * (1.f / NN) + LNEPS);
  for (int n = tid; n < NN; n += 256) {
    float y = (buf[n] - mu) * inv * g[n] + be[n];
    out[(long)G * NN + n] = y;
    buf[n] = y;
  }
  __syncthreads();
  for (int pi = tid; pi < 576; pi += 256) {
    int p_y = pi / 24, p_x = pi % 24;
    int base = (p_y * 2) * WW + p_x * 2;
    float m4 = fmaxf(fmaxf(buf[base], buf[base + 1]), fmaxf(buf[base + WW], buf[base + WW + 1]));
    pool[(long)G * 576 + pi] = m4;
  }
}

// ---------------- host orchestration ----------------
extern "C" void kernel_launch(void* const* d_in, const int* in_sizes, int n_in,
                              void* d_out, int out_size, void* d_ws, size_t ws_size,
                              hipStream_t stream) {
  const float* x       = (const float*)d_in[0];
  const float* conv1_w = (const float*)d_in[1];
  const float* conv1_b = (const float*)d_in[2];
  const float* conv2_w = (const float*)d_in[3];
  const float* conv2_b = (const float*)d_in[4];
  const float* id_w    = (const float*)d_in[5];
  const float* id_b    = (const float*)d_in[6];
  const float* q_w     = (const float*)d_in[7];
  const float* q_b     = (const float*)d_in[8];
  const float* k_w     = (const float*)d_in[9];
  const float* k_b     = (const float*)d_in[10];
  const float* v_w     = (const float*)d_in[11];
  const float* v_b     = (const float*)d_in[12];
  const float* rel_t   = (const float*)d_in[13];
  const float* ln1_g   = (const float*)d_in[14];
  const float* ln1_b   = (const float*)d_in[15];
  const float* ln2_g   = (const float*)d_in[16];
  const float* ln2_b   = (const float*)d_in[17];
  const float* ffn_w1  = (const float*)d_in[18];
  const float* ffn_b1  = (const float*)d_in[19];
  const float* ffn_w2  = (const float*)d_in[20];
  const float* ffn_b2  = (const float*)d_in[21];

  // workspace arena (~830 MB), 256B aligned chunks
  char* p = (char*)d_ws;
  auto take = [&](size_t bytes) -> char* {
    char* r = p; p += (bytes + 255) & ~(size_t)255; return r;
  };
  bf16_t* w2t   = (bf16_t*)take((size_t)9 * CC * CC * 2);
  bf16_t* qw_b  = (bf16_t*)take((size_t)EE * CC * 2);
  bf16_t* kw_b  = (bf16_t*)take((size_t)EE * CC * 2);
  bf16_t* vw_b  = (bf16_t*)take((size_t)EE * CC * 2);
  bf16_t* w1_b  = (bf16_t*)take((size_t)EE * 2 * EE * 2);
  bf16_t* w2f_b = (bf16_t*)take((size_t)2 * EE * EE * 2);
  bf16_t* hpad  = (bf16_t*)take((size_t)BB * CC * PADC * 2);
  float*  down_f = (float*)take((size_t)BB * CC * NN * 4);   // also reused as FFN output f
  bf16_t* down_b = (bf16_t*)take((size_t)BB * CC * NN * 2);
  bf16_t* q_bf   = (bf16_t*)take((size_t)BB * EE * NN * 2);
  bf16_t* kT     = (bf16_t*)take((size_t)BB * NN * EE * 2);
  bf16_t* v_bf   = (bf16_t*)take((size_t)BB * EE * NN * 2);
  float*  qk_f   = (float*)take((size_t)BB * EE * EE * 4);   // also reused as ao
  bf16_t* attn_b = (bf16_t*)take((size_t)BB * EE * EE * 2);
  float*  h1_f   = (float*)take((size_t)BB * CC * NN * 4);
  bf16_t* h1_b   = (bf16_t*)take((size_t)BB * CC * NN * 2);
  bf16_t* f1_b   = (bf16_t*)take((size_t)BB * CC * 2 * EE * 2);
  float*  ao_f = qk_f;     // lifetime-disjoint alias
  float*  f_f  = down_f;   // lifetime-disjoint alias

  const dim3 blk(256);

  // weight conversion / repack
  { long n = (long)9 * CC * CC;
    repack_conv2<<<dim3((unsigned)((n + 255) / 256)), blk, 0, stream>>>(conv2_w, w2t); }
  { long n = (long)EE * CC;
    f32_to_bf16<<<dim3((unsigned)((n + 255) / 256)), blk, 0, stream>>>(q_w, qw_b, n);
    f32_to_bf16<<<dim3((unsigned)((n + 255) / 256)), blk, 0, stream>>>(k_w, kw_b, n);
    f32_to_bf16<<<dim3((unsigned)((n + 255) / 256)), blk, 0, stream>>>(v_w, vw_b, n); }
  { long n = (long)EE * 2 * EE;
    f32_to_bf16<<<dim3((unsigned)((n + 255) / 256)), blk, 0, stream>>>(ffn_w1, w1_b, n);
    f32_to_bf16<<<dim3((unsigned)((n + 255) / 256)), blk, 0, stream>>>(ffn_w2, w2f_b, n); }

  // conv1 + relu into padded bf16 image
  { long n = (long)BB * CC * 2500;
    conv1_relu_pad<<<dim3((unsigned)((n + 255) / 256)), blk, 0, stream>>>(x, conv1_w, conv1_b, hpad); }

  // conv2 + identity residual + relu -> down (fp32 + bf16)
  conv2_res_gemm<<<dim3(9, 72, BB), blk, 0, stream>>>(w2t, hpad, conv2_b, id_w, id_b, x, down_f, down_b);

  // q / k / v projections (k stored transposed for q·k^T)
  gemm_bf16_wmma<<<dim3(9, 72, BB), blk, 0, stream>>>(
      qw_b, CC, 0, down_b, NN, (long)CC * NN, EE, NN, CC, 1.f,
      q_b, nullptr, 0, nullptr, q_bf, NN, (long)EE * NN,
      nullptr, 0, 0, nullptr, 0);
  gemm_bf16_wmma<<<dim3(9, 72, BB), blk, 0, stream>>>(
      kw_b, CC, 0, down_b, NN, (long)CC * NN, EE, NN, CC, 1.f,
      k_b, nullptr, 0, nullptr, nullptr, NN, (long)EE * NN,
      kT, EE, (long)NN * EE, nullptr, 0);
  gemm_bf16_wmma<<<dim3(9, 72, BB), blk, 0, stream>>>(
      vw_b, CC, 0, down_b, NN, (long)CC * NN, EE, NN, CC, 1.f,
      v_b, nullptr, 0, nullptr, v_bf, NN, (long)EE * NN,
      nullptr, 0, 0, nullptr, 0);

  // q·k^T / sqrt(C) + relative position bias -> qk (fp32)
  gemm_bf16_wmma<<<dim3(9, 72, BB), blk, 0, stream>>>(
      q_bf, NN, (long)EE * NN, kT, EE, (long)NN * EE, EE, EE, NN, 1.f / 48.f,
      nullptr, nullptr, 0, qk_f, nullptr, EE, (long)EE * EE,
      nullptr, 0, 0, rel_t, 1);

  // softmax -> attn (bf16)
  softmax_rows<<<dim3(BB * EE), blk, 0, stream>>>(qk_f, attn_b);

  // attn · v -> ao (fp32, aliases qk buffer)
  gemm_bf16_wmma<<<dim3(9, 72, BB), blk, 0, stream>>>(
      attn_b, EE, (long)EE * EE, v_bf, NN, (long)EE * NN, EE, NN, EE, 1.f,
      nullptr, nullptr, 0, ao_f, nullptr, NN, (long)EE * NN,
      nullptr, 0, 0, nullptr, 0);

  // permute(1,0,2)-gather + residual + LN1 -> h1 (fp32 + bf16)
  ln1_residual<<<dim3(BB * CC), blk, 0, stream>>>(ao_f, down_f, ln1_g, ln1_b, h1_f, h1_b);

  // FFN: h1 @ w1 (+b1, relu) -> f1 (bf16) ; f1 @ w2 (+b2) -> f (fp32, aliases down_f)
  gemm_bf16_wmma<<<dim3(18, 72, BB), blk, 0, stream>>>(
      h1_b, NN, (long)CC * NN, w1_b, 2 * EE, 0, CC, 2 * EE, NN, 1.f,
      nullptr, ffn_b1, 1, nullptr, f1_b, 2 * EE, (long)CC * 2 * EE,
      nullptr, 0, 0, nullptr, 0);
  gemm_bf16_wmma<<<dim3(9, 72, BB), blk, 0, stream>>>(
      f1_b, 2 * EE, (long)CC * 2 * EE, w2f_b, EE, 0, CC, EE, 2 * EE, 1.f,
      nullptr, ffn_b2, 0, f_f, nullptr, EE, (long)CC * EE,
      nullptr, 0, 0, nullptr, 0);

  // residual + LN2 + fused 2x2 maxpool -> d_out (out | pool)
  float* outp  = (float*)d_out;
  float* poolp = outp + (size_t)BB * CC * NN;
  ln2_pool<<<dim3(BB * CC), blk, 0, stream>>>(f_f, h1_f, ln2_g, ln2_b, outp, poolp);
}
